// FLGNN_90941637525593
// MI455X (gfx1250) — matrix operand order repaired and checked by reading, not compiled, not run on hardware
//
#include <hip/hip_runtime.h>
#include <hip/hip_bf16.h>

#define NN   32768
#define EE   1048576
#define INC  128
#define HC   128
#define OUTC 40
#define LL   3
#define RR   16
#define NPAD 48   // OUT padded to 3 WMMA tiles

typedef __attribute__((ext_vector_type(16))) __bf16 bfrag_t;
typedef __attribute__((ext_vector_type(8)))  float  facc_t;

union FragU { bfrag_t v; unsigned int u[8]; };

__device__ __forceinline__ __bf16 f2bf(float f) { return (__bf16)f; }

__device__ __forceinline__ facc_t facc_zero() {
    facc_t z = {0.f,0.f,0.f,0.f,0.f,0.f,0.f,0.f};
    return z;
}

// ---- CDNA5 async global->LDS DMA (no VGPR staging; tracked by ASYNCcnt) ----
// LDS byte address from a VGPR; global address is a 64-bit SGPR base plus a 32-bit VGPR offset
__device__ __forceinline__ void async_copy_b128(unsigned lds_off, const void* gbase,
                                                unsigned byte_off) {
    asm volatile("global_load_async_to_lds_b128 %0, %1, %2"
                 :
                 : "v"(lds_off), "v"(byte_off), "s"(gbase)
                 : "memory");
}

__device__ __forceinline__ void wait_async0() {
    asm volatile("s_wait_asynccnt 0x0" ::: "memory");
}

// Generic pointer to a __shared__ object keeps the LDS byte offset in its
// low 32 bits (addrspacecast 3->generic = {aperture_hi, lds_off}).
__device__ __forceinline__ unsigned lds_addr(const void* shared_ptr) {
    return (unsigned)(unsigned long long)shared_ptr;
}

// A fragment (16x32 bf16, M x K), rows m0..m0+15, cols k0..k0+31.
// LDS row-major [rows][ldk] bf16. ISA layout: lanes 0-15 K-groups {0-7,16-23},
// lanes 16-31 K-groups {8-15,24-31}; pairs of consecutive K per dword.
__device__ __forceinline__ bfrag_t load_frag_A(const __bf16* base, int m0, int k0,
                                               int ldk, int lane) {
    int m    = m0 + (lane & 15);
    int half = lane >> 4;
    const unsigned int* p = (const unsigned int*)(base + m * ldk);
    FragU f;
#pragma unroll
    for (int j = 0; j < 8; ++j) {
        int k = k0 + ((j >> 2) << 4) + (half << 3) + ((j & 3) << 1);
        f.u[j] = p[k >> 1];
    }
    return f.v;
}

// B fragment (32x16 bf16, K x N) from transposed LDS tile BT[n][ldk].
// lanes 0-15: K-groups {0-7,16-23}, lanes 16-31: K-groups {8-15,24-31} (pairs per dword).
__device__ __forceinline__ bfrag_t load_frag_B(const __bf16* baseT, int n0, int k0,
                                               int ldk, int lane) {
    int n    = n0 + (lane & 15);
    int half = lane >> 4;
    const unsigned int* p = (const unsigned int*)(baseT + n * ldk);
    FragU f;
#pragma unroll
    for (int j = 0; j < 8; ++j) {
        int k = k0 + ((j >> 2) << 4) + (half << 3) + ((j & 3) << 1);
        f.u[j] = p[k >> 1];
    }
    return f.v;
}

__device__ __forceinline__ facc_t wmma_bf16(bfrag_t a, bfrag_t b, facc_t c) {
    return __builtin_amdgcn_wmma_f32_16x16x32_bf16(false, a, false, b,
                                                   (short)0, c, false, false);
}

// ---------------------------------------------------------------- prep kernels

// W_in [IN][H] f32 -> W_inT [H][IN] bf16
__global__ void k_cvt_winT(const float* __restrict__ W, __bf16* __restrict__ BT) {
    int idx = blockIdx.x * 256 + threadIdx.x;           // [o][k]
    int o = idx >> 7, k = idx & 127;
    BT[idx] = f2bf(W[k * HC + o]);
}

// Wc [L*R][H][O] f32 -> WcT [L*R][O][H] bf16
__global__ void k_cvt_wcT(const float* __restrict__ W, __bf16* __restrict__ BT) {
    int idx = blockIdx.x * 256 + threadIdx.x;           // dest linear
    int lr  = idx >> 14;
    int rem = idx & 16383;
    int o = rem >> 7, hh = rem & 127;
    BT[idx] = f2bf(W[lr * (HC * HC) + hh * HC + o]);
}

// W_head [H][OUT] f32 -> W_headT [NPAD][H] bf16 (zero padded)
__global__ void k_cvt_whT(const float* __restrict__ W, __bf16* __restrict__ BT) {
    int idx = blockIdx.x * 256 + threadIdx.x;
    int o = idx >> 7, k = idx & 127;
    BT[idx] = f2bf(o < OUTC ? W[k * OUTC + o] : 0.f);
}

// inv2w2 = 0.5 / widths^2
__global__ void k_inv2w2(const float* __restrict__ w, float* __restrict__ inv) {
    int idx = blockIdx.x * 256 + threadIdx.x;
    float v = w[idx];
    inv[idx] = 0.5f / (v * v);
}

// ------------------------------------------------------------------ degree

__global__ void k_deg(const int* __restrict__ dst, float* __restrict__ deg) {
    int e = blockIdx.x * 256 + threadIdx.x;
    atomicAdd(&deg[dst[e]], 1.0f);
}

// ---------------------------------------------------- h0 = relu(x@W_in + b_in)

__global__ void k_gemm_in(const float* __restrict__ x, const __bf16* __restrict__ BT,
                          const float* __restrict__ bias, float* __restrict__ h) {
    __shared__ __bf16 sA[64 * HC];
    __shared__ __bf16 sB[HC * HC];
    int tid = threadIdx.x, wid = tid >> 5, lane = tid & 31;
    int m0g = blockIdx.x * 64;
    // B tile: async DMA straight into LDS (32 KB = 8 x b128 per thread)
    {
        unsigned sb = lds_addr(&sB[0]);
#pragma unroll
        for (int i = 0; i < 8; ++i) {
            unsigned off = (unsigned)(tid + i * 256) * 16u;
            async_copy_b128(sb + off, BT, off);
        }
    }
#pragma unroll
    for (int i = 0; i < 32; ++i) {
        int idx = tid + i * 256;
        sA[idx] = f2bf(x[m0g * HC + idx]);
    }
    wait_async0();
    __syncthreads();

    int n0 = wid * 16, half = lane >> 4, coll = lane & 15;
    facc_t y[4] = {facc_zero(), facc_zero(), facc_zero(), facc_zero()};
#pragma unroll
    for (int ks = 0; ks < 4; ++ks) {
        bfrag_t b = load_frag_B(sB, n0, ks * 32, HC, lane);
#pragma unroll
        for (int mt = 0; mt < 4; ++mt) {
            bfrag_t a = load_frag_A(sA, mt * 16, ks * 32, HC, lane);
            y[mt] = wmma_bf16(a, b, y[mt]);
        }
    }
    int col = n0 + coll;
    float bv = bias[col];
#pragma unroll
    for (int mt = 0; mt < 4; ++mt)
#pragma unroll
        for (int v = 0; v < 8; ++v) {
            int row = m0g + mt * 16 + half * 8 + v;
            h[row * HC + col] = fmaxf(y[mt][v] + bv, 0.f);
        }
}

// -------------------------------------------- rule memberships -> w [N,R]

__global__ void k_member(const float* __restrict__ h, const float* __restrict__ centers,
                         const float* __restrict__ inv2, float* __restrict__ w, int l) {
    int wid = threadIdx.x >> 5, lane = threadIdx.x & 31;
    int n = blockIdx.x * 8 + wid;
    float4 hv = ((const float4*)(h + n * HC))[lane];
    const float* cb = centers + l * RR * HC;
    const float* ib = inv2 + l * RR * HC;
    float s[RR];
#pragma unroll
    for (int r = 0; r < RR; ++r) {
        float4 cv = ((const float4*)(cb + r * HC))[lane];
        float4 iv = ((const float4*)(ib + r * HC))[lane];
        float dx = hv.x - cv.x, dy = hv.y - cv.y, dz = hv.z - cv.z, dw = hv.w - cv.w;
        float p = dx * dx * iv.x + dy * dy * iv.y + dz * dz * iv.z + dw * dw * iv.w;
        p += __shfl_xor(p, 16); p += __shfl_xor(p, 8); p += __shfl_xor(p, 4);
        p += __shfl_xor(p, 2);  p += __shfl_xor(p, 1);
        s[r] = p * (-1.0f / HC);            // logit = -mean(d2)
    }
    float m = s[0];
#pragma unroll
    for (int r = 1; r < RR; ++r) m = fmaxf(m, s[r]);
    float sum = 0.f;
#pragma unroll
    for (int r = 0; r < RR; ++r) { s[r] = __expf(s[r] - m); sum += s[r]; }
    float invs = 1.0f / sum;
    if (lane < RR) w[n * RR + lane] = s[lane] * invs;
}

// ------------- msg = sum_r w[:,r] * (h @ Wc[l,r] + bc[l,r])  (bf16 WMMA)

__global__ void k_msg(const float* __restrict__ h, const __bf16* __restrict__ wcT_l,
                      const float* __restrict__ bc_l, const float* __restrict__ w,
                      float* __restrict__ msg) {
    __shared__ __bf16 sA[64 * HC];      // h tile      16 KB
    __shared__ __bf16 sB[HC * HC];      // WcT[r] tile 32 KB
    __shared__ float  sW[64 * RR];      // firing str.  4 KB
    int tid = threadIdx.x, wid = tid >> 5, lane = tid & 31;
    int m0g = blockIdx.x * 64;
#pragma unroll
    for (int i = 0; i < 32; ++i) {
        int idx = tid + i * 256;
        sA[idx] = f2bf(h[m0g * HC + idx]);
    }
#pragma unroll
    for (int i = 0; i < 4; ++i) {
        int idx = tid + i * 256;
        sW[idx] = w[m0g * RR + idx];
    }
    int n0 = wid * 16, half = lane >> 4, coll = lane & 15;
    unsigned sb = lds_addr(&sB[0]);
    facc_t acc[4] = {facc_zero(), facc_zero(), facc_zero(), facc_zero()};

    for (int r = 0; r < RR; ++r) {
        __syncthreads();                 // prior rule's LDS reads retired
        // Stage WcT[l,r] (32 KB) via async DMA, no VGPR round-trip.
        {
            const char* gsrc = (const char*)(wcT_l + (size_t)r * HC * HC);
#pragma unroll
            for (int i = 0; i < 8; ++i) {
                unsigned off = (unsigned)(tid + i * 256) * 16u;
                async_copy_b128(sb + off, gsrc, off);
            }
        }
        wait_async0();
        __syncthreads();
        // Prefetch next rule's tile toward cache while we compute this one.
        if (r + 1 < RR)
            __builtin_prefetch(wcT_l + (size_t)(r + 1) * HC * HC + tid * 64, 0, 1);

        float bcr = bc_l[r * HC + n0 + coll];
        facc_t y[4] = {facc_zero(), facc_zero(), facc_zero(), facc_zero()};
#pragma unroll
        for (int ks = 0; ks < 4; ++ks) {
            bfrag_t b = load_frag_B(sB, n0, ks * 32, HC, lane);
#pragma unroll
            for (int mt = 0; mt < 4; ++mt) {
                bfrag_t a = load_frag_A(sA, mt * 16, ks * 32, HC, lane);
                y[mt] = wmma_bf16(a, b, y[mt]);
            }
        }
#pragma unroll
        for (int mt = 0; mt < 4; ++mt)
#pragma unroll
            for (int v = 0; v < 8; ++v) {
                float wv = sW[(mt * 16 + half * 8 + v) * RR + r];
                acc[mt][v] += wv * (y[mt][v] + bcr);
            }
    }
    int col = n0 + coll;
#pragma unroll
    for (int mt = 0; mt < 4; ++mt)
#pragma unroll
        for (int v = 0; v < 8; ++v) {
            int row = m0g + mt * 16 + half * 8 + v;
            msg[row * HC + col] = acc[mt][v];
        }
}

// ----------------------------------------------- edge scatter (agg += msg[src])

__global__ void k_scatter(const int* __restrict__ src, const int* __restrict__ dst,
                          const float* __restrict__ msg, float* __restrict__ agg) {
    int idx = blockIdx.x * 256 + threadIdx.x;
    int e = idx >> 7, c = idx & 127;
    atomicAdd(&agg[dst[e] * HC + c], msg[src[e] * HC + c]);
}

// ----------------- a = relu(agg/deg); channel sums for BN; a stored to abuf

__global__ void k_bnstat(const float* __restrict__ agg, const float* __restrict__ deg,
                         float* __restrict__ abuf, float* __restrict__ bnsum) {
    int tid = threadIdx.x;
    int c = tid & 127, sub = tid >> 7;
    int base = blockIdx.x * 64 + sub * 32;
    float s = 0.f, s2 = 0.f;
#pragma unroll 4
    for (int i = 0; i < 32; ++i) {
        int n = base + i;
        float d = fmaxf(deg[n], 1.0f);
        float a = fmaxf(agg[n * HC + c] / d, 0.f);
        abuf[n * HC + c] = a;
        s += a; s2 += a * a;
    }
    __shared__ float sS[256], sQ[256];
    sS[tid] = s; sQ[tid] = s2;
    __syncthreads();
    if (sub == 0) {
        s  = sS[tid] + sS[tid + 128];
        s2 = sQ[tid] + sQ[tid + 128];
        atomicAdd(&bnsum[c], s);
        atomicAdd(&bnsum[HC + c], s2);
    }
}

__global__ void k_bnfin(const float* __restrict__ bnsum, const float* __restrict__ gamma,
                        const float* __restrict__ beta, float* __restrict__ bnsc, int l) {
    int c = threadIdx.x;
    float mu  = bnsum[c] * (1.0f / NN);
    float var = fmaxf(bnsum[HC + c] * (1.0f / NN) - mu * mu, 0.f);
    float sc  = gamma[l * HC + c] * rsqrtf(var + 1e-5f);
    bnsc[c] = sc;
    bnsc[HC + c] = beta[l * HC + c] - mu * sc;
}

__global__ void k_hupd(float* __restrict__ h, const float* __restrict__ abuf,
                       const float* __restrict__ bnsc) {
    int idx = blockIdx.x * 256 + threadIdx.x;
    int c = idx & 127;
    h[idx] += abuf[idx] * bnsc[c] + bnsc[HC + c];
}

// ------------------------------- out = softmax(h @ W_head + b_head) over 40

__global__ void k_head(const float* __restrict__ h, const __bf16* __restrict__ whT,
                       const float* __restrict__ bhead, float* __restrict__ out) {
    __shared__ __bf16 sA[128 * HC];     // 32 KB
    __shared__ __bf16 sB[NPAD * HC];    // 12 KB
    __shared__ float  sY[128 * NPAD];   // 24 KB
    int tid = threadIdx.x, wid = tid >> 5, lane = tid & 31;
    int m0g = blockIdx.x * 128;
    {
        unsigned sb = lds_addr(&sB[0]);
#pragma unroll
        for (int i = 0; i < 3; ++i) {
            unsigned off = (unsigned)(tid + i * 256) * 16u;
            async_copy_b128(sb + off, whT, off);
        }
    }
#pragma unroll
    for (int i = 0; i < 64; ++i) {
        int idx = tid + i * 256;
        sA[idx] = f2bf(h[m0g * HC + idx]);
    }
    wait_async0();
    __syncthreads();
    int half = lane >> 4, coll = lane & 15;
#pragma unroll
    for (int nt = 0; nt < 3; ++nt) {
        facc_t y = facc_zero();
#pragma unroll
        for (int ks = 0; ks < 4; ++ks) {
            bfrag_t b = load_frag_B(sB, nt * 16, ks * 32, HC, lane);
            bfrag_t a = load_frag_A(sA, wid * 16, ks * 32, HC, lane);
            y = wmma_bf16(a, b, y);
        }
        int col = nt * 16 + coll;
        float bv = (col < OUTC) ? bhead[col] : 0.f;
#pragma unroll
        for (int v = 0; v < 8; ++v)
            sY[(wid * 16 + half * 8 + v) * NPAD + col] = y[v] + bv;
    }
    __syncthreads();
    if (tid < 128) {
        int row = tid;
        float m = -1e30f;
        for (int o = 0; o < OUTC; ++o) m = fmaxf(m, sY[row * NPAD + o]);
        float s = 0.f;
        for (int o = 0; o < OUTC; ++o) {
            float e = __expf(sY[row * NPAD + o] - m);
            sY[row * NPAD + o] = e; s += e;
        }
        float inv = 1.0f / s;
        for (int o = 0; o < OUTC; ++o)
            out[(m0g + row) * OUTC + o] = sY[row * NPAD + o] * inv;
    }
}

// --------------------------------------------------------------------- launch

extern "C" void kernel_launch(void* const* d_in, const int* in_sizes, int n_in,
                              void* d_out, int out_size, void* d_ws, size_t ws_size,
                              hipStream_t stream) {
    const float* x       = (const float*)d_in[0];
    const int*   ei      = (const int*)d_in[1];
    const int*   esrc    = ei;
    const int*   edst    = ei + EE;
    const float* W_in    = (const float*)d_in[2];
    const float* b_in    = (const float*)d_in[3];
    const float* centers = (const float*)d_in[4];
    const float* widths  = (const float*)d_in[5];
    const float* Wc      = (const float*)d_in[6];
    const float* bc      = (const float*)d_in[7];
    const float* gamma   = (const float*)d_in[8];
    const float* beta    = (const float*)d_in[9];
    const float* W_head  = (const float*)d_in[10];
    const float* b_head  = (const float*)d_in[11];
    float* out = (float*)d_out;

    char* p = (char*)d_ws;
    auto carve = [&](size_t bytes) -> char* {
        char* q = p;
        p += (bytes + 255) & ~(size_t)255;
        return q;
    };
    float*  h     = (float*)carve((size_t)NN * HC * 4);
    float*  msg   = (float*)carve((size_t)NN * HC * 4);   // reused as abuf post-scatter
    float*  agg   = (float*)carve((size_t)NN * HC * 4);
    float*  wbuf  = (float*)carve((size_t)NN * RR * 4);
    float*  deg   = (float*)carve((size_t)NN * 4);
    float*  bnsum = (float*)carve(2 * HC * 4);
    float*  bnsc  = (float*)carve(2 * HC * 4);
    __bf16* winT  = (__bf16*)carve((size_t)HC * INC * 2);
    __bf16* wcT   = (__bf16*)carve((size_t)LL * RR * HC * HC * 2);
    __bf16* whT   = (__bf16*)carve((size_t)NPAD * HC * 2);
    float*  inv2  = (float*)carve((size_t)LL * RR * HC * 4);

    hipMemsetAsync(deg, 0, (size_t)NN * 4, stream);
    k_cvt_winT<<<(HC * INC) / 256, 256, 0, stream>>>(W_in, winT);
    k_cvt_wcT<<<(LL * RR * HC * HC) / 256, 256, 0, stream>>>(Wc, wcT);
    k_cvt_whT<<<(NPAD * HC) / 256, 256, 0, stream>>>(W_head, whT);
    k_inv2w2<<<(LL * RR * HC) / 256, 256, 0, stream>>>(widths, inv2);

    k_deg<<<EE / 256, 256, 0, stream>>>(edst, deg);
    k_gemm_in<<<NN / 64, 256, 0, stream>>>(x, winT, b_in, h);

    for (int l = 0; l < LL; ++l) {
        k_member<<<NN / 8, 256, 0, stream>>>(h, centers, inv2, wbuf, l);
        k_msg<<<NN / 64, 256, 0, stream>>>(h, wcT + (size_t)l * RR * HC * HC,
                                           bc + (size_t)l * RR * HC, wbuf, msg);
        hipMemsetAsync(agg, 0, (size_t)NN * HC * 4, stream);
        k_scatter<<<(EE * HC) / 256, 256, 0, stream>>>(esrc, edst, msg, agg);
        hipMemsetAsync(bnsum, 0, 2 * HC * 4, stream);
        k_bnstat<<<NN / 64, 256, 0, stream>>>(agg, deg, msg /*abuf*/, bnsum);
        k_bnfin<<<1, 128, 0, stream>>>(bnsum, gamma, beta, bnsc, l);
        k_hupd<<<(NN * HC) / 256, 256, 0, stream>>>(h, msg, bnsc);
    }
    k_head<<<NN / 128, 256, 0, stream>>>(h, whT, b_head, out);
}
